// NeighborhoodAttention1D_16406775071494
// MI455X (gfx1250) — matrix operand order repaired and checked by reading, not compiled, not run on hardware
//
#include <hip/hip_runtime.h>
#include <hip/hip_bf16.h>

// ---------------------------------------------------------------------------
// NeighborhoodAttention1D for MI455X (gfx1250, wave32, WMMA, async-to-LDS)
//   qkv = x @ w_qkv + b_qkv          -> bf16 WMMA GEMM, f32 accum
//   windowed attention (kk=13)        -> wave-per-(b,h,l), shfl reductions
//   out = attn_out @ w_proj + b_proj  -> bf16 WMMA GEMM, f32 accum
// ---------------------------------------------------------------------------

static constexpr int BB = 2;
static constexpr int LL = 4096;
static constexpr int CC = 1024;
static constexpr int HH = 16;
static constexpr int DD = 64;   // CC / HH
static constexpr int MAXK = 19;

typedef __attribute__((ext_vector_type(8)))  __bf16 v8bf;
typedef __attribute__((ext_vector_type(16))) __bf16 v16bf;
typedef __attribute__((ext_vector_type(8)))  float  v8f;
typedef __attribute__((ext_vector_type(4)))  int    v4i;

union V16 { v16bf v; v8bf h[2]; };

#ifndef __has_builtin
#define __has_builtin(x) 0
#endif

#if __has_builtin(__builtin_amdgcn_global_load_async_to_lds_b128)
#define USE_ASYNC_COPY 1
#else
#define USE_ASYNC_COPY 0
#endif

#define GLOBAL_AS __attribute__((address_space(1)))
#define LDS_AS    __attribute__((address_space(3)))

// Issue one 16B global->LDS copy (async engine when available).
__device__ __forceinline__ void copy16_g2l(const __bf16* g, __bf16* l) {
#if USE_ASYNC_COPY
    __builtin_amdgcn_global_load_async_to_lds_b128(
        (GLOBAL_AS v4i*)(GLOBAL_AS void*)g,
        (LDS_AS    v4i*)(LDS_AS    void*)l, 0, 0);
#else
    *(v8bf*)l = *(const v8bf*)g;
#endif
}

__device__ __forceinline__ void async_copy_wait() {
#if USE_ASYNC_COPY
#if __has_builtin(__builtin_amdgcn_s_wait_asynccnt)
    __builtin_amdgcn_s_wait_asynccnt(0);
#else
    asm volatile("s_wait_asynccnt 0x0" ::: "memory");
#endif
#endif
}

// ----------------------------- f32 -> bf16 ---------------------------------
__global__ __launch_bounds__(256) void cvt_f32_bf16(const float* __restrict__ in,
                                                    __bf16* __restrict__ out, int n) {
    int i = (blockIdx.x * 256 + threadIdx.x) * 4;
    if (i + 3 < n) {
        float4 f = *(const float4*)&in[i];
        out[i + 0] = (__bf16)f.x;
        out[i + 1] = (__bf16)f.y;
        out[i + 2] = (__bf16)f.z;
        out[i + 3] = (__bf16)f.w;
    }
}

// ------------------------- bf16 WMMA tiled GEMM ----------------------------
// C[M,N] = A[M,K] @ B[K,N] + bias[N]   (A,B bf16 row-major; C f32 row-major)
// Block tile 128x128, BK=32, 256 threads = 8 waves (4 along M x 2 along N).
// Each wave computes 32x64 = 2x4 grid of 16x16 WMMA tiles (8 accumulators).
// Double-buffered LDS fed by async global->LDS copies.
#define GBM 128
#define GBN 128
#define GBK 32

__global__ __launch_bounds__(256) void gemm_bf16_wmma(
        const __bf16* __restrict__ A, const __bf16* __restrict__ Bm,
        const float* __restrict__ bias, float* __restrict__ Cout,
        int M, int N, int K) {
    __shared__ __bf16 As[2][GBM][GBK];
    __shared__ __bf16 Bs[2][GBK][GBN];

    const int tid   = threadIdx.x;
    const int lane  = tid & 31;
    const int wv    = tid >> 5;     // 0..7
    const int waveM = wv & 3;       // 4 waves along M (32 rows each)
    const int waveN = wv >> 2;      // 2 waves along N (64 cols each)
    const int hi    = lane >> 4;    // half-wave select
    const int m16   = lane & 15;

    const int blockM = blockIdx.y * GBM;
    const int blockN = blockIdx.x * GBN;

    // Per-thread staging assignment: 4x 16B chunks per K-step.
    // A tile 128x32 = 512 chunks (4 per row); B tile 32x128 = 512 chunks (16 per row).
    const int ac0_row = (tid      ) >> 2, ac0_col = ((tid      ) & 3)  << 3;
    const int ac1_row = (tid + 256) >> 2, ac1_col = ((tid + 256) & 3)  << 3;
    const int bc0_row = (tid      ) >> 4, bc0_col = ((tid      ) & 15) << 3;
    const int bc1_row = (tid + 256) >> 4, bc1_col = ((tid + 256) & 15) << 3;

    v8f acc[2][4] = {};

    // Prologue: stage K-step 0 into buffer 0.
    copy16_g2l(&A[(size_t)(blockM + ac0_row) * K + ac0_col], &As[0][ac0_row][ac0_col]);
    copy16_g2l(&A[(size_t)(blockM + ac1_row) * K + ac1_col], &As[0][ac1_row][ac1_col]);
    copy16_g2l(&Bm[(size_t)bc0_row * N + blockN + bc0_col],  &Bs[0][bc0_row][bc0_col]);
    copy16_g2l(&Bm[(size_t)bc1_row * N + blockN + bc1_col],  &Bs[0][bc1_row][bc1_col]);
    async_copy_wait();
    __syncthreads();

    int p = 0;
    for (int k0 = 0; k0 < K; k0 += GBK) {
        // Prefetch next K-step into the other buffer (async, no wait yet).
        if (k0 + GBK < K) {
            const int q = p ^ 1;
            const int kn = k0 + GBK;
            copy16_g2l(&A[(size_t)(blockM + ac0_row) * K + kn + ac0_col], &As[q][ac0_row][ac0_col]);
            copy16_g2l(&A[(size_t)(blockM + ac1_row) * K + kn + ac1_col], &As[q][ac1_row][ac1_col]);
            copy16_g2l(&Bm[(size_t)(kn + bc0_row) * N + blockN + bc0_col], &Bs[q][bc0_row][bc0_col]);
            copy16_g2l(&Bm[(size_t)(kn + bc1_row) * N + blockN + bc1_col], &Bs[q][bc1_row][bc1_col]);
        }

        // --- gather WMMA fragments from LDS buffer p ---
        // A 16x32 bf16: hi=0 lanes hold K 0..7 / 16..23, hi=1 lanes K 8..15 / 24..31.
        V16 afrag[2], bfrag[4];
        #pragma unroll
        for (int mt = 0; mt < 2; ++mt) {
            int row = waveM * 32 + mt * 16 + m16;
            afrag[mt].h[0] = *(const v8bf*)&As[p][row][hi * 8];
            afrag[mt].h[1] = *(const v8bf*)&As[p][row][hi * 8 + 16];
        }
        // B 32x16 bf16: lane column N = ncol (lane&15); hi=0 lanes hold K 0..7 / 16..23,
        //               hi=1 lanes K 8..15 / 24..31 (same order as A).
        #pragma unroll
        for (int nt = 0; nt < 4; ++nt) {
            int ncol = waveN * 64 + nt * 16 + m16;
            #pragma unroll
            for (int j = 0; j < 8; ++j) {
                bfrag[nt].v[j]     = Bs[p][hi * 8 + j][ncol];
                bfrag[nt].v[j + 8] = Bs[p][hi * 8 + 16 + j][ncol];
            }
        }

        #pragma unroll
        for (int mt = 0; mt < 2; ++mt)
            #pragma unroll
            for (int nt = 0; nt < 4; ++nt)
                acc[mt][nt] = __builtin_amdgcn_wmma_f32_16x16x32_bf16(
                    false, afrag[mt].v, false, bfrag[nt].v,
                    (short)0, acc[mt][nt], false, false);

        // Ensure prefetch landed and all waves are done reading buffer p.
        async_copy_wait();
        __syncthreads();
        p ^= 1;
    }

    // --- writeback: C/D layout: VGPR r -> M = r + 8*hi, N = lane&15 ---
    #pragma unroll
    for (int mt = 0; mt < 2; ++mt)
        #pragma unroll
        for (int nt = 0; nt < 4; ++nt) {
            int col = blockN + waveN * 64 + nt * 16 + m16;
            float bcol = bias[col];
            #pragma unroll
            for (int r = 0; r < 8; ++r) {
                int row = blockM + waveM * 32 + mt * 16 + hi * 8 + r;
                Cout[(size_t)row * N + col] = acc[mt][nt][r] + bcol;
            }
        }
}

// --------------------------- windowed attention -----------------------------
// One wave32 per (b,h,l). qkv f32 laid out (B,L,3,H,D). Each lane owns d,d+1.
__global__ __launch_bounds__(256) void neigh_attn_kernel(
        const float* __restrict__ qkv, const float* __restrict__ rpb,
        const int* __restrict__ knp, __bf16* __restrict__ attn_out) {
    const int kn = *knp;
    int r = 0;
    while ((r + 1) * (r + 1) <= kn) ++r;   // isqrt
    const int kk = r + 1;                  // 13 for kn=144

    const int wid  = (blockIdx.x * 256 + threadIdx.x) >> 5;   // 0 .. B*H*L-1
    const int lane = threadIdx.x & 31;
    const int l = wid & (LL - 1);
    const int h = (wid >> 12) & (HH - 1);
    const int b = wid >> 16;

    int start = l - (kk >> 1);
    if (start < 0) start = 0;
    if (start > LL - kk) start = LL - kk;

    const size_t rowq = ((size_t)(b * LL + l) * 3 + 0) * CC + h * DD + 2 * lane;
    const float q0 = qkv[rowq];
    const float q1 = qkv[rowq + 1];
    const float scale = 0.125f;            // D^-0.5, D=64

    float att[MAXK];
    #pragma unroll 1
    for (int j = 0; j < kk; ++j) {
        const size_t rowk = ((size_t)(b * LL + start + j) * 3 + 1) * CC + h * DD + 2 * lane;
        float s = q0 * qkv[rowk] + q1 * qkv[rowk + 1];
        #pragma unroll
        for (int off = 16; off > 0; off >>= 1)
            s += __shfl_xor(s, off, 32);
        att[j] = s * scale + rpb[h * (2 * kk - 1) + (start + j - l + kk - 1)];
    }

    float mx = att[0];
    for (int j = 1; j < kk; ++j) mx = fmaxf(mx, att[j]);
    float sum = 0.0f;
    for (int j = 0; j < kk; ++j) { att[j] = __expf(att[j] - mx); sum += att[j]; }
    const float inv = 1.0f / sum;

    float o0 = 0.0f, o1 = 0.0f;
    #pragma unroll 1
    for (int j = 0; j < kk; ++j) {
        const size_t rowv = ((size_t)(b * LL + start + j) * 3 + 2) * CC + h * DD + 2 * lane;
        o0 += att[j] * qkv[rowv];
        o1 += att[j] * qkv[rowv + 1];
    }
    const size_t oidx = (size_t)(b * LL + l) * CC + h * DD + 2 * lane;
    attn_out[oidx]     = (__bf16)(o0 * inv);
    attn_out[oidx + 1] = (__bf16)(o1 * inv);
}

// ------------------------------- launcher -----------------------------------
extern "C" void kernel_launch(void* const* d_in, const int* in_sizes, int n_in,
                              void* d_out, int out_size, void* d_ws, size_t ws_size,
                              hipStream_t stream) {
    const float* x      = (const float*)d_in[0];
    const float* w_qkv  = (const float*)d_in[1];
    const float* b_qkv  = (const float*)d_in[2];
    const float* rpb    = (const float*)d_in[3];
    const float* w_proj = (const float*)d_in[4];
    const float* b_proj = (const float*)d_in[5];
    const int*   kn     = (const int*)d_in[6];
    float* out = (float*)d_out;

    const size_t nX  = (size_t)BB * LL * CC;       // 8.4M
    const size_t nWq = (size_t)CC * 3 * CC;        // 3.1M
    const size_t nWp = (size_t)CC * CC;            // 1.0M
    const size_t nQKV = (size_t)BB * LL * 3 * CC;  // 25.2M

    char* ws = (char*)d_ws;
    __bf16* x_bf    = (__bf16*)ws;                    ws += nX  * sizeof(__bf16);
    __bf16* wqkv_bf = (__bf16*)ws;                    ws += nWq * sizeof(__bf16);
    __bf16* wproj_bf= (__bf16*)ws;                    ws += nWp * sizeof(__bf16);
    float*  qkv     = (float*)ws;                     ws += nQKV * sizeof(float);
    __bf16* attn_bf = (__bf16*)ws;                    ws += nX  * sizeof(__bf16);

    // f32 -> bf16 operand conversions
    cvt_f32_bf16<<<(int)(nX  / 1024), 256, 0, stream>>>(x,      x_bf,     (int)nX);
    cvt_f32_bf16<<<(int)(nWq / 1024), 256, 0, stream>>>(w_qkv,  wqkv_bf,  (int)nWq);
    cvt_f32_bf16<<<(int)(nWp / 1024), 256, 0, stream>>>(w_proj, wproj_bf, (int)nWp);

    // qkv = x @ w_qkv + b_qkv   : M=8192, N=3072, K=1024
    {
        dim3 grid(3 * CC / GBN, BB * LL / GBM);
        gemm_bf16_wmma<<<grid, 256, 0, stream>>>(x_bf, wqkv_bf, b_qkv, qkv,
                                                 BB * LL, 3 * CC, CC);
    }

    // windowed attention: B*H*L waves, 8 waves per block
    {
        int nwaves = BB * HH * LL;             // 131072
        neigh_attn_kernel<<<nwaves / 8, 256, 0, stream>>>(qkv, rpb, kn, attn_bf);
    }

    // out = attn @ w_proj + b_proj : M=8192, N=1024, K=1024
    {
        dim3 grid(CC / GBN, BB * LL / GBM);
        gemm_bf16_wmma<<<grid, 256, 0, stream>>>(attn_bf, wproj_bf, b_proj, out,
                                                 BB * LL, CC, CC);
    }
}
